// HMM_VAE_67001489817710
// MI455X (gfx1250) — compile-verified, matrix operand order checked by reading, not run on hardware
//
#include <hip/hip_runtime.h>
#include <hip/hip_bf16.h>

typedef __attribute__((ext_vector_type(16))) _Float16 v16h;
typedef __attribute__((ext_vector_type(8)))  _Float16 v8h;
typedef __attribute__((ext_vector_type(8)))  float    v8f;

#define KMIX 10
#define LATD 32
#define BQ   32
#define TQ   64
#define BT   2048          // BQ*TQ
#define KBT  20480         // KMIX*BT
#define D1   3136          // 64*7*7 encoder feature dim
#define DEC  512

// ---------------------------------------------------------------------------
// WMMA helpers: 16x16 tile per wave32, f16 inputs, f32 accum.
// ISA 16-bit A/B 16-rows x 32-K layout: lane<16 -> K {k0..k0+7, k0+16..k0+23},
//                                       lane>=16 -> K {k0+8..k0+15, k0+24..k0+31}.
// ---------------------------------------------------------------------------
__device__ __forceinline__ v16h load_k32_tile(const _Float16* __restrict__ p,
                                              int row0, int ld, int k0, int lane) {
  const int hl = lane >> 4;       // half selector
  const int r  = lane & 15;       // row within tile
  const _Float16* q = p + (size_t)(row0 + r) * (size_t)ld + (size_t)(k0 + hl * 8);
  union { v16h v; v8h h[2]; } u;
  u.h[0] = *(const v8h*)(q);        // K offsets +0..7  (or +8..15)
  u.h[1] = *(const v8h*)(q + 16);   // K offsets +16..23 (or +24..31)
  return u.v;
}

__device__ __forceinline__ v8f wmma_f16(v16h a, v16h b, v8f c) {
  return __builtin_amdgcn_wmma_f32_16x16x32_f16(
      /*neg_a=*/false, a, /*neg_b=*/false, b,
      /*c_mod=*/(short)0, c, /*reuse_a=*/false, /*reuse_b=*/false);
}

// ---------------------------------------------------------------------------
// Weight conversion: fce_w (64x3136), Wz = fcd1_w[:, :32] (512x32, packed),
// fcd2_w (3136x512) -> f16, K-contiguous, 16B-aligned strides.
// ---------------------------------------------------------------------------
__global__ void convert_weights_kernel(const float* __restrict__ fce_w,
                                       const float* __restrict__ fcd1_w,
                                       const float* __restrict__ fcd2_w,
                                       _Float16* __restrict__ fceh,
                                       _Float16* __restrict__ wzh,
                                       _Float16* __restrict__ fcd2h) {
  int i = blockIdx.x * 256 + threadIdx.x;
  if (i < 200704) { fceh[i] = (_Float16)fce_w[i]; return; }
  int i2 = i - 200704;
  if (i2 < 16384) {                       // Wz: gather (512 x 32) from (512 x 42)
    int n = i2 >> 5, k = i2 & 31;
    wzh[i2] = (_Float16)fcd1_w[n * 42 + k];
    return;
  }
  int i3 = i2 - 16384;
  if (i3 < 1605632) fcd2h[i3] = (_Float16)fcd2_w[i3];
}

// ---------------------------------------------------------------------------
// Encoder conv1: (BT,1,28,28) -> relu -> (BT,32,14,14), k=4 s=2 p=1
// ---------------------------------------------------------------------------
__global__ void conv1_kernel(const float* __restrict__ y,
                             const float* __restrict__ w,
                             const float* __restrict__ b,
                             float* __restrict__ out) {
  int idx = blockIdx.x * 256 + threadIdx.x;       // BT*32*196
  int ox = idx % 14; int t = idx / 14;
  int oy = t % 14;   t /= 14;
  int oc = t % 32;   int bt = t / 32;
  if (bt >= BT) return;
  const float* img = y + (size_t)bt * 784;
  const float* wk  = w + oc * 16;
  float acc = b[oc];
#pragma unroll
  for (int ky = 0; ky < 4; ++ky) {
    int iy = oy * 2 - 1 + ky;
    if (iy < 0 || iy >= 28) continue;
#pragma unroll
    for (int kx = 0; kx < 4; ++kx) {
      int ix = ox * 2 - 1 + kx;
      if (ix < 0 || ix >= 28) continue;
      acc += img[iy * 28 + ix] * wk[ky * 4 + kx];
    }
  }
  out[idx] = fmaxf(acc, 0.f);
}

// ---------------------------------------------------------------------------
// Encoder conv2: (BT,32,14,14) -> relu -> h f16 (BT, 3136)
// ---------------------------------------------------------------------------
__global__ void conv2_kernel(const float* __restrict__ in,
                             const float* __restrict__ w,
                             const float* __restrict__ b,
                             _Float16* __restrict__ hh) {
  int idx = blockIdx.x * 256 + threadIdx.x;       // BT*64*49
  int ox = idx % 7; int t = idx / 7;
  int oy = t % 7;   t /= 7;
  int oc = t % 64;  int bt = t / 64;
  if (bt >= BT) return;
  const float* ibase = in + (size_t)bt * 32 * 196;
  float acc = b[oc];
  for (int ic = 0; ic < 32; ++ic) {
    const float* ip = ibase + ic * 196;
    const float* wk = w + ((oc * 32 + ic) << 4);
#pragma unroll
    for (int ky = 0; ky < 4; ++ky) {
      int iy = oy * 2 - 1 + ky;
      if (iy < 0 || iy >= 14) continue;
#pragma unroll
      for (int kx = 0; kx < 4; ++kx) {
        int ix = ox * 2 - 1 + kx;
        if (ix < 0 || ix >= 14) continue;
        acc += ip[iy * 14 + ix] * wk[ky * 4 + kx];
      }
    }
  }
  hh[idx] = (_Float16)fmaxf(acc, 0.f);
}

// ---------------------------------------------------------------------------
// GEMM 1: enc = h (2048x3136,f16) x fce_w^T (3136x64) + fce_b -> f32.
// One wave computes a 16x64 strip (whole N), A fragment reused across 4 WMMAs.
// ---------------------------------------------------------------------------
__global__ void gemm_enc_kernel(const _Float16* __restrict__ A,
                                const _Float16* __restrict__ Bm,
                                const float* __restrict__ bias,
                                float* __restrict__ out) {
  int lane = threadIdx.x & 31, wave = threadIdx.x >> 5;
  int mt = blockIdx.x * 8 + wave;     // 128 M-tiles
  int m0 = mt * 16;
  v8f a0 = {}, a1 = {}, a2 = {}, a3 = {};
  for (int k0 = 0; k0 < D1; k0 += 32) {
    if (k0 + 32 < D1)
      __builtin_prefetch(A + (size_t)m0 * D1 + k0 + 32, 0, 1);
    v16h a  = load_k32_tile(A, m0, D1, k0, lane);
    v16h b0 = load_k32_tile(Bm, 0,  D1, k0, lane);
    v16h b1 = load_k32_tile(Bm, 16, D1, k0, lane);
    v16h b2 = load_k32_tile(Bm, 32, D1, k0, lane);
    v16h b3 = load_k32_tile(Bm, 48, D1, k0, lane);
    a0 = wmma_f16(a, b0, a0);
    a1 = wmma_f16(a, b1, a1);
    a2 = wmma_f16(a, b2, a2);
    a3 = wmma_f16(a, b3, a3);
  }
  int hl = lane >> 4, r = lane & 15;
  union { v8f v; float f[8]; } u0, u1, u2, u3;
  u0.v = a0; u1.v = a1; u2.v = a2; u3.v = a3;
#pragma unroll
  for (int j = 0; j < 8; ++j) {
    int m = m0 + j + 8 * hl;
    out[(size_t)m * 64 + r]      = u0.f[j] + bias[r];
    out[(size_t)m * 64 + 16 + r] = u1.f[j] + bias[16 + r];
    out[(size_t)m * 64 + 32 + r] = u2.f[j] + bias[32 + r];
    out[(size_t)m * 64 + 48 + r] = u3.f[j] + bias[48 + r];
  }
}

// ---------------------------------------------------------------------------
// Reparameterize: z = mu + eps*exp(0.5*logvar) -> f16 (2048x32)
// ---------------------------------------------------------------------------
__global__ void reparam_kernel(const float* __restrict__ enc,
                               const float* __restrict__ eps,
                               _Float16* __restrict__ zh) {
  int idx = blockIdx.x * 256 + threadIdx.x;       // BT*32
  int l = idx & 31, bt = idx >> 5;
  if (bt >= BT) return;
  float mu = enc[bt * 64 + l];
  float lv = enc[bt * 64 + 32 + l];
  zh[idx] = (_Float16)(mu + eps[idx] * expf(0.5f * lv));
}

// ---------------------------------------------------------------------------
// GEMM 2: base = z (2048x32,f16) x Wz^T (32x512) + fcd1_b -> f32.
// K=32: one WMMA per tile; wave does a 16x64 strip (4 tiles, A reused).
// ---------------------------------------------------------------------------
__global__ void gemm_base_kernel(const _Float16* __restrict__ A,
                                 const _Float16* __restrict__ Bm,
                                 const float* __restrict__ bias,
                                 float* __restrict__ out) {
  int lane = threadIdx.x & 31, wave = threadIdx.x >> 5;
  int id = blockIdx.x * 8 + wave;     // 1024 jobs = 128 Mt x 8 Ngroups
  int mt = id >> 3, ng = id & 7;
  int m0 = mt * 16, n0 = ng * 64;
  v16h a  = load_k32_tile(A, m0, 32, 0, lane);
  v16h b0 = load_k32_tile(Bm, n0,      32, 0, lane);
  v16h b1 = load_k32_tile(Bm, n0 + 16, 32, 0, lane);
  v16h b2 = load_k32_tile(Bm, n0 + 32, 32, 0, lane);
  v16h b3 = load_k32_tile(Bm, n0 + 48, 32, 0, lane);
  v8f z = {};
  v8f a0 = wmma_f16(a, b0, z);
  v8f a1 = wmma_f16(a, b1, z);
  v8f a2 = wmma_f16(a, b2, z);
  v8f a3 = wmma_f16(a, b3, z);
  int hl = lane >> 4, r = lane & 15;
  union { v8f v; float f[8]; } u0, u1, u2, u3;
  u0.v = a0; u1.v = a1; u2.v = a2; u3.v = a3;
#pragma unroll
  for (int j = 0; j < 8; ++j) {
    int m = m0 + j + 8 * hl;
    out[(size_t)m * DEC + n0 + r]      = u0.f[j] + bias[n0 + r];
    out[(size_t)m * DEC + n0 + 16 + r] = u1.f[j] + bias[n0 + 16 + r];
    out[(size_t)m * DEC + n0 + 32 + r] = u2.f[j] + bias[n0 + 32 + r];
    out[(size_t)m * DEC + n0 + 48 + r] = u3.f[j] + bias[n0 + 48 + r];
  }
}

// ---------------------------------------------------------------------------
// h1[k,m,d] = relu(base[m,d] + fcd1_w[d, 32+k]) -> f16 (KBT x 512)
// ---------------------------------------------------------------------------
__global__ void h1_kernel(const float* __restrict__ base,
                          const float* __restrict__ fcd1_w,
                          _Float16* __restrict__ h1h) {
  int idx = blockIdx.x * 256 + threadIdx.x;       // KMIX*BT*512
  int d = idx % DEC; int t = idx / DEC;
  int m = t % BT;    int k = t / BT;
  if (k >= KMIX) return;
  float v = base[(size_t)m * DEC + d] + fcd1_w[d * 42 + 32 + k];
  h1h[idx] = (_Float16)fmaxf(v, 0.f);
}

// ---------------------------------------------------------------------------
// GEMM 3 (the big one): h2 = h1 (20480x512,f16) x fcd2_w^T (512x3136)
//                        + fcd2_b, relu -> f16 (20480x3136)
// Block = 8 waves = 128 rows of M, all sharing one 64-column N-group.
// B slab (64x32 f16 = 4KB) staged in LDS once per k-step; each wave does a
// 16x64 strip with 4 accumulators (A fragment reused across 4 WMMAs).
// ---------------------------------------------------------------------------
__global__ void gemm_h2_kernel(const _Float16* __restrict__ A,
                               const _Float16* __restrict__ Bm,
                               const float* __restrict__ bias,
                               _Float16* __restrict__ out) {
  __shared__ __attribute__((aligned(16))) _Float16 bs[64 * 32];   // 4 KB
  int tid  = threadIdx.x;
  int lane = tid & 31, wave = tid >> 5;
  int n0 = blockIdx.x * 64;                       // 49 N-groups
  int m0 = (blockIdx.y * 8 + wave) * 16;          // 1280 M-tiles
  int brow = tid >> 2, bchunk = (tid & 3) * 8;    // 64 rows x 4 chunks of 8 halves
  v8f a0 = {}, a1 = {}, a2 = {}, a3 = {};
  for (int k0 = 0; k0 < DEC; k0 += 32) {
    __syncthreads();                              // protect previous-iter reads
    *(v8h*)(bs + brow * 32 + bchunk) =
        *(const v8h*)(Bm + (size_t)(n0 + brow) * DEC + k0 + bchunk);
    __syncthreads();
    if (k0 + 32 < DEC)
      __builtin_prefetch(A + (size_t)m0 * DEC + k0 + 32, 0, 1);
    v16h a  = load_k32_tile(A, m0, DEC, k0, lane);
    v16h b0 = load_k32_tile(bs,           0, 32, 0, lane);
    v16h b1 = load_k32_tile(bs + 16 * 32, 0, 32, 0, lane);
    v16h b2 = load_k32_tile(bs + 32 * 32, 0, 32, 0, lane);
    v16h b3 = load_k32_tile(bs + 48 * 32, 0, 32, 0, lane);
    a0 = wmma_f16(a, b0, a0);
    a1 = wmma_f16(a, b1, a1);
    a2 = wmma_f16(a, b2, a2);
    a3 = wmma_f16(a, b3, a3);
  }
  int hl = lane >> 4, r = lane & 15;
  union { v8f v; float f[8]; } u0, u1, u2, u3;
  u0.v = a0; u1.v = a1; u2.v = a2; u3.v = a3;
  float bn0 = bias[n0 + r],      bn1 = bias[n0 + 16 + r];
  float bn2 = bias[n0 + 32 + r], bn3 = bias[n0 + 48 + r];
#pragma unroll
  for (int j = 0; j < 8; ++j) {
    int m = m0 + j + 8 * hl;
    _Float16* o = out + (size_t)m * D1 + n0 + r;
    o[0]  = (_Float16)fmaxf(u0.f[j] + bn0, 0.f);
    o[16] = (_Float16)fmaxf(u1.f[j] + bn1, 0.f);
    o[32] = (_Float16)fmaxf(u2.f[j] + bn2, 0.f);
    o[48] = (_Float16)fmaxf(u3.f[j] + bn3, 0.f);
  }
}

// ---------------------------------------------------------------------------
// deconv1: (KBT,64,7,7) f16 -> relu -> (KBT,32,14,14) f16, k=4 s=2 p=1
// y[n,oc,oy,ox] = b + sum_ic sum_{ky,kx: 2*iy-1+ky==oy} x[n,ic,iy,ix]*w[ic,oc,ky,kx]
// ---------------------------------------------------------------------------
__global__ void deconv1_kernel(const _Float16* __restrict__ x,
                               const float* __restrict__ w,
                               const float* __restrict__ b,
                               _Float16* __restrict__ out) {
  int idx = blockIdx.x * 256 + threadIdx.x;       // KBT*32*196
  int ox = idx % 14; int t = idx / 14;
  int oy = t % 14;   t /= 14;
  int oc = t % 32;   int n = t / 32;
  if (n >= KBT) return;
  const _Float16* ibase = x + (size_t)n * D1;
  float acc = b[oc];
#pragma unroll
  for (int ky = 0; ky < 4; ++ky) {
    int ty = oy + 1 - ky;
    if (ty & 1) continue;
    int iy = ty >> 1;
    if (iy < 0 || iy >= 7) continue;
#pragma unroll
    for (int kx = 0; kx < 4; ++kx) {
      int tx = ox + 1 - kx;
      if (tx & 1) continue;
      int ix = tx >> 1;
      if (ix < 0 || ix >= 7) continue;
      const _Float16* ip = ibase + iy * 7 + ix;
      const float*    wk = w + oc * 16 + ky * 4 + kx;   // w[ic,oc,ky,kx], ic stride 512
      for (int ic = 0; ic < 64; ++ic)
        acc += (float)ip[ic * 49] * wk[ic * 512];
    }
  }
  out[idx] = (_Float16)fmaxf(acc, 0.f);
}

// ---------------------------------------------------------------------------
// deconv2 fused with sigmoid + BCE reduction: one block per (k,bt) image.
// bce[n] = sum_p -( t*max(log p,-100) + (1-t)*max(log(1-p),-100) )
// ---------------------------------------------------------------------------
__global__ void deconv2_bce_kernel(const _Float16* __restrict__ x,
                                   const float* __restrict__ w,
                                   const float* __restrict__ b,
                                   const float* __restrict__ y,
                                   float* __restrict__ bce) {
  __shared__ float red[256];
  int n  = blockIdx.x;                 // 0..KBT-1 ; n = k*BT + bt
  int bt = n % BT;
  const _Float16* ibase = x + (size_t)n * 32 * 196;
  const float* tgt = y + (size_t)bt * 784;
  float bias = b[0];
  float s = 0.f;
  for (int p = threadIdx.x; p < 784; p += 256) {
    int oy = p / 28, ox = p % 28;
    float acc = bias;
#pragma unroll
    for (int ky = 0; ky < 4; ++ky) {
      int ty = oy + 1 - ky;
      if (ty & 1) continue;
      int iy = ty >> 1;
      if (iy < 0 || iy >= 14) continue;
#pragma unroll
      for (int kx = 0; kx < 4; ++kx) {
        int tx = ox + 1 - kx;
        if (tx & 1) continue;
        int ix = tx >> 1;
        if (ix < 0 || ix >= 14) continue;
        const _Float16* ip = ibase + iy * 14 + ix;
        const float*    wk = w + ky * 4 + kx;        // dc2_w (32,1,4,4): ic stride 16
        for (int ic = 0; ic < 32; ++ic)
          acc += (float)ip[ic * 196] * wk[ic * 16];
      }
    }
    float pr = 1.f / (1.f + expf(-acc));
    float tg = tgt[p];
    float lp = fmaxf(logf(pr), -100.f);
    float lq = fmaxf(logf(1.f - pr), -100.f);
    s += -(tg * lp + (1.f - tg) * lq);
  }
  red[threadIdx.x] = s;
  __syncthreads();
  for (int off = 128; off > 0; off >>= 1) {
    if (threadIdx.x < off) red[threadIdx.x] += red[threadIdx.x + off];
    __syncthreads();
  }
  if (threadIdx.x == 0) bce[n] = red[0];
}

// ---------------------------------------------------------------------------
// HMM forward (logsumexp scan) + backward gumbel-softmax sampling.
// One thread per batch sequence b (K=10, T=64: tiny).
// ---------------------------------------------------------------------------
__global__ void hmm_kernel(const float* __restrict__ bce,
                           const float* __restrict__ gum,
                           const float* __restrict__ log_pi,
                           const float* __restrict__ log_A,
                           float* __restrict__ alpha_ws,
                           float* __restrict__ out) {
  int b = threadIdx.x;
  if (b >= BQ) return;

  float logA[KMIX][KMIX], logpi[KMIX];
  for (int i = 0; i < KMIX; ++i) {
    float m = -1e30f;
    for (int j = 0; j < KMIX; ++j) m = fmaxf(m, log_A[i * KMIX + j]);
    float sum = 0.f;
    for (int j = 0; j < KMIX; ++j) sum += expf(log_A[i * KMIX + j] - m);
    for (int j = 0; j < KMIX; ++j)
      logA[i][j] = logf(expf(log_A[i * KMIX + j] - m) / sum + 1e-9f);
  }
  {
    float m = -1e30f;
    for (int j = 0; j < KMIX; ++j) m = fmaxf(m, log_pi[j]);
    float sum = 0.f;
    for (int j = 0; j < KMIX; ++j) sum += expf(log_pi[j] - m);
    for (int j = 0; j < KMIX; ++j)
      logpi[j] = logf(expf(log_pi[j] - m) / sum + 1e-9f);
  }

  float* al = alpha_ws + (size_t)b * TQ * KMIX;
  // log_bk[b,t,k] = -bce[k*BT + b*TQ + t] / 50
  for (int k = 0; k < KMIX; ++k)
    al[k] = logpi[k] - bce[k * BT + b * TQ] * 0.02f;
  for (int t = 1; t < TQ; ++t) {
    for (int k = 0; k < KMIX; ++k) {
      float m = -1e30f;
      for (int j = 0; j < KMIX; ++j)
        m = fmaxf(m, al[(t - 1) * KMIX + j] + logA[j][k]);
      float s = 0.f;
      for (int j = 0; j < KMIX; ++j)
        s += expf(al[(t - 1) * KMIX + j] + logA[j][k] - m);
      al[t * KMIX + k] = m + logf(s) - bce[k * BT + b * TQ + t] * 0.02f;
    }
  }

  float xn[KMIX];
  {
    float m = -1e30f;
    for (int k = 0; k < KMIX; ++k) m = fmaxf(m, al[(TQ - 1) * KMIX + k]);
    float e[KMIX], s = 0.f;
    for (int k = 0; k < KMIX; ++k) {
      e[k] = expf(al[(TQ - 1) * KMIX + k] - m + gum[(b * TQ + TQ - 1) * KMIX + k]);
      s += e[k];
    }
    for (int k = 0; k < KMIX; ++k) {
      xn[k] = e[k] / s;
      out[(b * TQ + TQ - 1) * KMIX + k] = xn[k];
    }
  }
  for (int t = TQ - 2; t >= 0; --t) {
    float logits[KMIX];
    for (int i = 0; i < KMIX; ++i) {
      float a = al[t * KMIX + i];
      for (int j = 0; j < KMIX; ++j) a += xn[j] * logA[i][j];
      logits[i] = a;
    }
    float m = -1e30f;
    for (int k = 0; k < KMIX; ++k) m = fmaxf(m, logits[k]);
    float e[KMIX], s = 0.f;
    for (int k = 0; k < KMIX; ++k) {
      e[k] = expf(logits[k] - m + gum[(b * TQ + t) * KMIX + k]);
      s += e[k];
    }
    for (int k = 0; k < KMIX; ++k) {
      xn[k] = e[k] / s;
      out[(b * TQ + t) * KMIX + k] = xn[k];
    }
  }
}

// ---------------------------------------------------------------------------
extern "C" void kernel_launch(void* const* d_in, const int* in_sizes, int n_in,
                              void* d_out, int out_size, void* d_ws, size_t ws_size,
                              hipStream_t stream) {
  const float* y      = (const float*)d_in[0];
  const float* eps    = (const float*)d_in[1];
  const float* gum    = (const float*)d_in[2];
  const float* c1w    = (const float*)d_in[3];
  const float* c1b    = (const float*)d_in[4];
  const float* c2w    = (const float*)d_in[5];
  const float* c2b    = (const float*)d_in[6];
  const float* fce_w  = (const float*)d_in[7];
  const float* fce_b  = (const float*)d_in[8];
  const float* fcd1_w = (const float*)d_in[9];
  const float* fcd1_b = (const float*)d_in[10];
  const float* fcd2_w = (const float*)d_in[11];
  const float* fcd2_b = (const float*)d_in[12];
  const float* dc1w   = (const float*)d_in[13];
  const float* dc1b   = (const float*)d_in[14];
  const float* dc2w   = (const float*)d_in[15];
  const float* dc2b   = (const float*)d_in[16];
  const float* log_pi = (const float*)d_in[17];
  const float* log_A  = (const float*)d_in[18];
  float* out = (float*)d_out;
  (void)in_sizes; (void)n_in; (void)out_size; (void)ws_size;

  char* p = (char*)d_ws;
  auto alloc = [&](size_t bytes) -> char* {
    char* r = p;
    p += (bytes + 255) & ~(size_t)255;
    return r;
  };
  _Float16* fceh  = (_Float16*)alloc((size_t)64 * D1 * 2);
  _Float16* wzh   = (_Float16*)alloc((size_t)DEC * 32 * 2);
  _Float16* fcd2h = (_Float16*)alloc((size_t)D1 * DEC * 2);
  float*    c1o   = (float*)   alloc((size_t)BT * 32 * 196 * 4);
  _Float16* hh    = (_Float16*)alloc((size_t)BT * D1 * 2);
  float*    enc   = (float*)   alloc((size_t)BT * 64 * 4);
  _Float16* zh    = (_Float16*)alloc((size_t)BT * 32 * 2);
  float*    base_ = (float*)   alloc((size_t)BT * DEC * 4);
  _Float16* h1h   = (_Float16*)alloc((size_t)KBT * DEC * 2);
  _Float16* h2h   = (_Float16*)alloc((size_t)KBT * D1 * 2);
  _Float16* dh    = (_Float16*)alloc((size_t)KBT * 32 * 196 * 2);
  float*    bce   = (float*)   alloc((size_t)KBT * 4);
  float*    alw   = (float*)   alloc((size_t)BQ * TQ * KMIX * 4);

  convert_weights_kernel<<<7120, 256, 0, stream>>>(fce_w, fcd1_w, fcd2_w,
                                                   fceh, wzh, fcd2h);
  conv1_kernel<<<50176, 256, 0, stream>>>(y, c1w, c1b, c1o);
  conv2_kernel<<<25088, 256, 0, stream>>>(c1o, c2w, c2b, hh);
  gemm_enc_kernel<<<16, 256, 0, stream>>>(hh, fceh, fce_b, enc);
  reparam_kernel<<<256, 256, 0, stream>>>(enc, eps, zh);
  gemm_base_kernel<<<128, 256, 0, stream>>>(zh, wzh, fcd1_b, base_);
  h1_kernel<<<40960, 256, 0, stream>>>(base_, fcd1_w, h1h);
  gemm_h2_kernel<<<dim3(49, 160), 256, 0, stream>>>(h1h, fcd2h, fcd2_b, h2h);
  deconv1_kernel<<<501760, 256, 0, stream>>>(h2h, dc1w, dc1b, dh);
  deconv2_bce_kernel<<<20480, 256, 0, stream>>>(dh, dc2w, dc2b, y, bce);
  hmm_kernel<<<1, 32, 0, stream>>>(bce, gum, log_pi, log_A, alw, out);
}
